// GCNNet_28707561406525
// MI455X (gfx1250) — compile-verified, matrix operand order checked by reading, not run on hardware
//
#include <hip/hip_runtime.h>

// ---------------------------------------------------------------------------
// Types for CDNA5 WMMA (wave32): v_wmma_f32_16x16x32_bf16
// ---------------------------------------------------------------------------
typedef __bf16 bf16;
typedef bf16  v16bf __attribute__((ext_vector_type(16)));
typedef float v8f   __attribute__((ext_vector_type(8)));

// ---------------------------------------------------------------------------
// Problem constants (from reference)
// ---------------------------------------------------------------------------
static constexpr long NN   = 50000;   // nodes
static constexpr long EE   = 200000;  // edges
static constexpr long NGR  = 256;     // graphs / batch
static constexpr long DF   = 334;     // node feature dim
static constexpr long LSEQ = 13132;   // cell-mut sequence length

// conv tower geometry
static constexpr long P1 = LSEQ - 7;      // 13125
static constexpr long Q1 = P1 / 3;        // 4375
static constexpr long P2 = Q1 - 7;        // 4368
static constexpr long Q2 = P2 / 3;        // 1456
static constexpr long P3 = Q2 - 7;        // 1449
static constexpr long Q3 = P3 / 3;        // 483 ; 128*483 = 61824

// ---------------------------------------------------------------------------
// A-tile loaders for the unified WMMA GEMM
// ---------------------------------------------------------------------------
struct DenseLoader {
    const float* a;
    int lda;
    __device__ float operator()(int m, int k) const {
        return a[(long)m * lda + k];
    }
    // 8 consecutive k starting at k8 (contiguous in memory)
    __device__ void load8(int m, int k8, float v[8]) const {
        const float* p = a + (long)m * lda + k8;
        #pragma unroll
        for (int j = 0; j < 8; ++j) v[j] = p[j];
    }
};

// Implicit-GEMM (im2col) loader for Conv1d VALID, K=8.
// m -> (batch b, out position p); k -> (in-channel ic, tap kk)
struct ConvLoader {
    const float* in;   // [B][Cin][Lin] contiguous
    int Cin, Lin, Pout;
    __device__ float operator()(int m, int k) const {
        int b  = m / Pout;
        int p  = m - b * Pout;
        int ic = k >> 3;
        int kk = k & 7;
        return in[((long)b * Cin + ic) * Lin + p + kk];
    }
    // k8 is a multiple of 8 -> single (ic) channel, taps 0..7: contiguous
    __device__ void load8(int m, int k8, float v[8]) const {
        int b  = m / Pout;
        int p  = m - b * Pout;
        int ic = k8 >> 3;
        const float* q = in + ((long)b * Cin + ic) * Lin + p;
        #pragma unroll
        for (int j = 0; j < 8; ++j) v[j] = q[j];
    }
};

// packed 8 x bf16 <-> uint4 for single 16B LDS stores
union Pack8 {
    bf16  h[8];
    uint4 u;
};
union Frag {
    v16bf v;
    uint4 q[2];
};

// ---------------------------------------------------------------------------
// Unified bf16 WMMA GEMM:  C[M,N] = act( A[M,K] @ B[K,N] + bias )
//   btrans==1 : B element (k,n) = B[n*K + k]   (conv weights [oc][ic*8])
//   act: 0 = identity, 1 = relu
// Block tile 64x64, BK=32, 8 waves (wave32), each wave -> 16x32 strip.
// M/N boundaries handled by source clamping (OOB rows/cols feed only
// unstored C entries); only the K tail needs zero-fill (one guarded step).
// ---------------------------------------------------------------------------
template <typename ALd>
__global__ __launch_bounds__(256)
void gemm_bf16_wmma(ALd aload, const float* __restrict__ B,
                    const float* __restrict__ bias, float* __restrict__ C,
                    int M, int N, int K, int btrans, int act)
{
    __shared__ __align__(16) bf16 As[64][40];   // [m][k] (80B rows, 16B aligned)
    __shared__ __align__(16) bf16 Bs[64][40];   // [n][k] (transposed tile)

    const int tid  = threadIdx.x;
    const int lane = tid & 31;
    const int wave = tid >> 5;
    const int wr   = wave >> 1;        // 0..3 : 16-row strip
    const int wcc  = wave & 1;         // 0..1 : 32-col strip
    const int m0   = blockIdx.y * 64;
    const int n0   = blockIdx.x * 64;

    const int hf  = lane >> 4;         // lane half (K interleave select)
    const int l15 = lane & 15;

    v8f acc0 = {0.f, 0.f, 0.f, 0.f, 0.f, 0.f, 0.f, 0.f};
    v8f acc1 = acc0;

    const int lm = tid >> 2;           // 0..63 : tile row (A) / tile col (B)
    const int lk = (tid & 3) * 8;      // 0,8,16,24
    const int mc = min(m0 + lm, M - 1);   // clamped A row (branch-free staging)
    const int nc = min(n0 + lm, N - 1);   // clamped B col

    const bf16* ap  = &As[wr * 16 + l15][hf * 8];
    const bf16* bp0 = &Bs[wcc * 32 + l15][hf * 8];
    const bf16* bp1 = &Bs[wcc * 32 + 16 + l15][hf * 8];

    const int Ktail = K & 31;
    const int Kmain = K - Ktail;

    for (int k0 = 0; k0 < Kmain; k0 += 32) {
        const int kk = k0 + lk;

        // ---- branch-free A staging: 2x global_load_b128 -> cvt -> ds_store_b128
        {
            float v[8];
            aload.load8(mc, kk, v);
            Pack8 pk;
            #pragma unroll
            for (int j = 0; j < 8; ++j) pk.h[j] = (bf16)v[j];
            *(uint4*)&As[lm][lk] = pk.u;
        }
        // ---- branch-free B staging (transposed into Bs[n][k]) ----
        {
            float v[8];
            if (btrans) {
                const float* p = B + (long)nc * K + kk;
                #pragma unroll
                for (int j = 0; j < 8; ++j) v[j] = p[j];
            } else {
                #pragma unroll
                for (int j = 0; j < 8; ++j) v[j] = B[(long)(kk + j) * N + nc];
            }
            Pack8 pk;
            #pragma unroll
            for (int j = 0; j < 8; ++j) pk.h[j] = (bf16)v[j];
            *(uint4*)&Bs[lm][lk] = pk.u;
        }
        // prefetch next B tile chunk (global_prefetch_b8)
        if (k0 + 32 < K) {
            const long nb = btrans ? ((long)nc * K + (kk + 32))
                                   : ((long)(kk + 32) * N + nc);
            __builtin_prefetch(B + nb, 0, 3);
        }
        __syncthreads();

        // ---- fragments per ISA 16-bit layouts ----
        // A 16x32: lane<16 -> K {0..7,16..23}; lane>=16 -> K {8..15,24..31}
        Frag fa, fb0, fb1;
        fa.q[0]  = *(const uint4*)(ap);
        fa.q[1]  = *(const uint4*)(ap + 16);
        fb0.q[0] = *(const uint4*)(bp0);
        fb0.q[1] = *(const uint4*)(bp0 + 16);
        fb1.q[0] = *(const uint4*)(bp1);
        fb1.q[1] = *(const uint4*)(bp1 + 16);

        acc0 = __builtin_amdgcn_wmma_f32_16x16x32_bf16(
                   false, fa.v, false, fb0.v, (short)0, acc0, false, false);
        acc1 = __builtin_amdgcn_wmma_f32_16x16x32_bf16(
                   false, fa.v, false, fb1.v, (short)0, acc1, false, false);
        __syncthreads();
    }

    // ---- single guarded K-tail step (zero-fill k >= K) ----
    if (Ktail) {
        const int kk = Kmain + lk;
        #pragma unroll
        for (int j = 0; j < 8; ++j) {
            const int gk = kk + j;
            As[lm][lk + j] = (bf16)((gk < K) ? aload(mc, gk) : 0.0f);
            float v = 0.0f;
            if (gk < K)
                v = btrans ? B[(long)nc * K + gk] : B[(long)gk * N + nc];
            Bs[lm][lk + j] = (bf16)v;
        }
        __syncthreads();

        Frag fa, fb0, fb1;
        fa.q[0]  = *(const uint4*)(ap);
        fa.q[1]  = *(const uint4*)(ap + 16);
        fb0.q[0] = *(const uint4*)(bp0);
        fb0.q[1] = *(const uint4*)(bp0 + 16);
        fb1.q[0] = *(const uint4*)(bp1);
        fb1.q[1] = *(const uint4*)(bp1 + 16);

        acc0 = __builtin_amdgcn_wmma_f32_16x16x32_bf16(
                   false, fa.v, false, fb0.v, (short)0, acc0, false, false);
        acc1 = __builtin_amdgcn_wmma_f32_16x16x32_bf16(
                   false, fa.v, false, fb1.v, (short)0, acc1, false, false);
        __syncthreads();
    }

    // ---- epilogue: C/D layout lane<16 -> M=j, lane>=16 -> M=j+8 ----
    const int ncol0 = n0 + wcc * 32 + l15;
    const int ncol1 = ncol0 + 16;
    const int mbase = m0 + wr * 16 + 8 * hf;
    #pragma unroll
    for (int j = 0; j < 8; ++j) {
        const int gm = mbase + j;
        if (gm >= M) continue;
        if (ncol0 < N) {
            float v = acc0[j] + (bias ? bias[ncol0] : 0.0f);
            if (act) v = fmaxf(v, 0.0f);
            C[(long)gm * N + ncol0] = v;
        }
        if (ncol1 < N) {
            float v = acc1[j] + (bias ? bias[ncol1] : 0.0f);
            if (act) v = fmaxf(v, 0.0f);
            C[(long)gm * N + ncol1] = v;
        }
    }
}

// ---------------------------------------------------------------------------
// Support kernels
// ---------------------------------------------------------------------------
__global__ void fill_f32(float* p, float v, long n) {
    long i = (long)blockIdx.x * 256 + threadIdx.x;
    if (i < n) p[i] = v;
}

__global__ void deg_kernel(const int* __restrict__ dst, float* deg, int E) {
    int e = blockIdx.x * 256 + threadIdx.x;
    if (e < E) atomicAdd(&deg[dst[e]], 1.0f);
}

__global__ void rsqrt_kernel(float* d, int n) {
    int i = blockIdx.x * 256 + threadIdx.x;
    if (i < n) d[i] = rsqrtf(d[i]);
}

__global__ void enorm_kernel(const int* __restrict__ src, const int* __restrict__ dst,
                             const float* __restrict__ dis, float* enorm, int E) {
    int e = blockIdx.x * 256 + threadIdx.x;
    if (e < E) enorm[e] = dis[src[e]] * dis[dst[e]];
}

// agg[dst, f] += t[src, f] * enorm[e]
__global__ void scatter_kernel(const float* __restrict__ t, float* agg,
                               const int* __restrict__ src, const int* __restrict__ dst,
                               const float* __restrict__ enorm, int E, int F) {
    long i = (long)blockIdx.x * 256 + threadIdx.x;
    if (i >= (long)E * F) return;
    int e = (int)(i / F);
    int f = (int)(i - (long)e * F);
    atomicAdd(&agg[(long)dst[e] * F + f], t[(long)src[e] * F + f] * enorm[e]);
}

// h = relu(agg + t*dis^2 + b)   (out may alias t; index-wise in-place safe)
__global__ void gcn_update(const float* __restrict__ t, const float* __restrict__ agg,
                           const float* __restrict__ dis, const float* __restrict__ bias,
                           float* out, int Nn, int F) {
    long i = (long)blockIdx.x * 256 + threadIdx.x;
    if (i >= (long)Nn * F) return;
    int n = (int)(i / F);
    int f = (int)(i - (long)n * F);
    float d = dis[n];
    float v = agg[i] + t[i] * d * d + bias[f];
    out[i] = fmaxf(v, 0.0f);
}

// per-graph max pool; h >= 0 (post-relu) so int-compare atomicMax is exact
__global__ void segmax_kernel(const float* __restrict__ h, const int* __restrict__ batch,
                              float* g, int Nn, int F) {
    long i = (long)blockIdx.x * 256 + threadIdx.x;
    if (i >= (long)Nn * F) return;
    int n = (int)(i / F);
    int f = (int)(i - (long)n * F);
    atomicMax((int*)&g[(long)batch[n] * F + f], __float_as_int(h[i]));
}

// conv1: Cin=1, K=8, relu fused; out layout [(b*P+p)][C]
__global__ void conv1_kernel(const float* __restrict__ x, const float* __restrict__ w,
                             const float* __restrict__ bias, float* out,
                             int B, int L, int P, int C) {
    long i = (long)blockIdx.x * 256 + threadIdx.x;
    if (i >= (long)B * P * C) return;
    int c = (int)(i % C);
    long row = i / C;
    int b = (int)(row / P);
    int p = (int)(row - (long)b * P);
    const float* xp = &x[(long)b * L + p];
    float s = bias[c];
    #pragma unroll
    for (int k = 0; k < 8; ++k) s += xp[k] * w[c * 8 + k];
    out[row * (long)C + c] = fmaxf(s, 0.0f);
}

// MaxPool1d(3,3): in [(b*P+p)][C] -> out [b][c][Lp]
__global__ void pool3_kernel(const float* __restrict__ in, float* out,
                             int B, int C, int P, int Lp) {
    long i = (long)blockIdx.x * 256 + threadIdx.x;
    if (i >= (long)B * C * Lp) return;
    int p = (int)(i % Lp);
    long t = i / Lp;
    int c = (int)(t % C);
    int b = (int)(t / C);
    long base = ((long)b * P + 3 * (long)p) * C + c;
    float v = fmaxf(fmaxf(in[base], in[base + C]), in[base + 2 * C]);
    out[i] = v;
}

__global__ void concat_kernel(const float* __restrict__ a, const float* __restrict__ b,
                              float* out) {  // [256,128]|[256,128] -> [256,256]
    int i = blockIdx.x * 256 + threadIdx.x;
    if (i >= 256 * 256) return;
    int r = i >> 8, c = i & 255;
    out[i] = (c < 128) ? a[r * 128 + c] : b[r * 128 + (c - 128)];
}

__global__ void final_kernel(const float* __restrict__ f2, const float* __restrict__ ow,
                             const float* __restrict__ ob, float* out) {
    int r = blockIdx.x * blockDim.x + threadIdx.x;
    if (r >= 256) return;
    float s = ob[0];
    #pragma unroll 8
    for (int j = 0; j < 128; ++j) s += f2[r * 128 + j] * ow[j];
    out[r] = 1.0f / (1.0f + expf(-s));
}

// ---------------------------------------------------------------------------
// Host launcher
// ---------------------------------------------------------------------------
static inline dim3 g1(long n) { return dim3((unsigned)((n + 255) / 256)); }

extern "C" void kernel_launch(void* const* d_in, const int* in_sizes, int n_in,
                              void* d_out, int out_size, void* d_ws, size_t ws_size,
                              hipStream_t stream) {
    (void)in_sizes; (void)n_in; (void)out_size; (void)ws_size;

    // inputs (setup_inputs dict order)
    const float* x      = (const float*)d_in[0];
    const float* xcm    = (const float*)d_in[1];   // [256,1,13132]
    const int*   ei     = (const int*)  d_in[3];   // [2,E]
    const int*   batch  = (const int*)  d_in[4];
    const float* W1 = (const float*)d_in[5],  *b1 = (const float*)d_in[6];
    const float* W2 = (const float*)d_in[7],  *b2 = (const float*)d_in[8];
    const float* W3 = (const float*)d_in[9],  *b3 = (const float*)d_in[10];
    const float* fcg1_w = (const float*)d_in[11], *fcg1_b = (const float*)d_in[12];
    const float* fcg2_w = (const float*)d_in[13], *fcg2_b = (const float*)d_in[14];
    const float* cw1 = (const float*)d_in[15], *cb1 = (const float*)d_in[16];
    const float* cw2 = (const float*)d_in[17], *cb2 = (const float*)d_in[18];
    const float* cw3 = (const float*)d_in[19], *cb3 = (const float*)d_in[20];
    const float* fcxt_w = (const float*)d_in[21], *fcxt_b = (const float*)d_in[22];
    const float* fc1_w  = (const float*)d_in[23], *fc1_b  = (const float*)d_in[24];
    const float* fc2_w  = (const float*)d_in[25], *fc2_b  = (const float*)d_in[26];
    const float* out_w  = (const float*)d_in[27], *out_b  = (const float*)d_in[28];
    float* out = (float*)d_out;

    const int* srcI = ei;
    const int* dstI = ei + EE;

    // workspace carve-out
    size_t off = 0;
    auto alloc = [&](size_t bytes) {
        void* p = (char*)d_ws + off;
        off = (off + bytes + 255) & ~(size_t)255;
        return p;
    };
    float* dis   = (float*)alloc(NN * 4);
    float* enorm = (float*)alloc(EE * 4);
    float* gmax  = (float*)alloc(NGR * 1336 * 4);
    float* gfc1  = (float*)alloc(NGR * 1024 * 4);
    float* gvec  = (float*)alloc(NGR * 128 * 4);
    float* xtv   = (float*)alloc(NGR * 128 * 4);
    float* xcb   = (float*)alloc(NGR * 256 * 4);
    float* f1    = (float*)alloc(NGR * 1024 * 4);
    float* f2    = (float*)alloc(NGR * 128 * 4);
    float* R0    = (float*)alloc((size_t)NGR * P1 * 32 * 4);   // 430 MB
    float* R1    = (float*)alloc((size_t)NN * 1336 * 4);       // 267 MB

    auto dense = [&](const float* A, int lda, const float* B, const float* bias,
                     float* C, int M, int N, int K, int btrans, int act) {
        dim3 grid((unsigned)((N + 63) / 64), (unsigned)((M + 63) / 64));
        gemm_bf16_wmma<DenseLoader><<<grid, 256, 0, stream>>>(
            DenseLoader{A, lda}, B, bias, C, M, N, K, btrans, act);
    };
    auto convg = [&](const float* in, int Cin, int Lin, int Pout,
                     const float* B, const float* bias, float* C, int N) {
        int M = (int)NGR * Pout, K = Cin * 8;
        dim3 grid((unsigned)((N + 63) / 64), (unsigned)((M + 63) / 64));
        gemm_bf16_wmma<ConvLoader><<<grid, 256, 0, stream>>>(
            ConvLoader{in, Cin, Lin, Pout}, B, bias, C, M, N, K, /*btrans=*/1, /*relu*/1);
    };

    // ---- GCN normalization ----
    fill_f32<<<g1(NN), 256, 0, stream>>>(dis, 1.0f, NN);                 // self-loop
    deg_kernel<<<g1(EE), 256, 0, stream>>>(dstI, dis, (int)EE);
    rsqrt_kernel<<<g1(NN), 256, 0, stream>>>(dis, (int)NN);
    enorm_kernel<<<g1(EE), 256, 0, stream>>>(srcI, dstI, dis, enorm, (int)EE);

    // ---- GCN layer 1 : t=R0, agg=R1, h1 -> R0 ----
    dense(x, (int)DF, W1, nullptr, R0, (int)NN, (int)DF, (int)DF, 0, 0);
    fill_f32<<<g1(NN * DF), 256, 0, stream>>>(R1, 0.0f, NN * DF);
    scatter_kernel<<<g1(EE * DF), 256, 0, stream>>>(R0, R1, srcI, dstI, enorm, (int)EE, (int)DF);
    gcn_update<<<g1(NN * DF), 256, 0, stream>>>(R0, R1, dis, b1, R0, (int)NN, (int)DF);

    // ---- GCN layer 2 : t=R1, agg=R0, h2 -> R1 ----
    dense(R0, (int)DF, W2, nullptr, R1, (int)NN, (int)(2 * DF), (int)DF, 0, 0);
    fill_f32<<<g1(NN * 2 * DF), 256, 0, stream>>>(R0, 0.0f, NN * 2 * DF);
    scatter_kernel<<<g1(EE * 2 * DF), 256, 0, stream>>>(R1, R0, srcI, dstI, enorm, (int)EE, (int)(2 * DF));
    gcn_update<<<g1(NN * 2 * DF), 256, 0, stream>>>(R1, R0, dis, b2, R1, (int)NN, (int)(2 * DF));

    // ---- GCN layer 3 : t=R0, agg=R1, h3 -> R0 ----
    dense(R1, (int)(2 * DF), W3, nullptr, R0, (int)NN, (int)(4 * DF), (int)(2 * DF), 0, 0);
    fill_f32<<<g1(NN * 4 * DF), 256, 0, stream>>>(R1, 0.0f, NN * 4 * DF);
    scatter_kernel<<<g1(EE * 4 * DF), 256, 0, stream>>>(R0, R1, srcI, dstI, enorm, (int)EE, (int)(4 * DF));
    gcn_update<<<g1(NN * 4 * DF), 256, 0, stream>>>(R0, R1, dis, b3, R0, (int)NN, (int)(4 * DF));

    // ---- global max pool + graph FC head ----
    fill_f32<<<g1(NGR * 1336), 256, 0, stream>>>(gmax, 0.0f, NGR * 1336);
    segmax_kernel<<<g1(NN * 1336), 256, 0, stream>>>(R0, batch, gmax, (int)NN, 1336);
    dense(gmax, 1336, fcg1_w, fcg1_b, gfc1, (int)NGR, 1024, 1336, 0, 1);
    dense(gfc1, 1024, fcg2_w, fcg2_b, gvec, (int)NGR, 128, 1024, 0, 0);

    // ---- conv tower (R0/R1 free now) ----
    conv1_kernel<<<g1(NGR * P1 * 32), 256, 0, stream>>>(xcm, cw1, cb1, R0,
                                                        (int)NGR, (int)LSEQ, (int)P1, 32);
    pool3_kernel<<<g1(NGR * 32 * Q1), 256, 0, stream>>>(R0, R1, (int)NGR, 32, (int)P1, (int)Q1);
    convg(R1, 32, (int)Q1, (int)P2, cw2, cb2, R0, 64);                     // implicit-GEMM conv2
    pool3_kernel<<<g1(NGR * 64 * Q2), 256, 0, stream>>>(R0, R1, (int)NGR, 64, (int)P2, (int)Q2);
    convg(R1, 64, (int)Q2, (int)P3, cw3, cb3, R0, 128);                    // implicit-GEMM conv3
    pool3_kernel<<<g1(NGR * 128 * Q3), 256, 0, stream>>>(R0, R1, (int)NGR, 128, (int)P3, (int)Q3);
    // R1 now holds flattened [256][61824] (flatten order = c*483+p, matches)
    dense(R1, 61824, fcxt_w, fcxt_b, xtv, (int)NGR, 128, 61824, 0, 0);

    // ---- fusion head ----
    concat_kernel<<<g1(NGR * 256), 256, 0, stream>>>(gvec, xtv, xcb);
    dense(xcb, 256, fc1_w, fc1_b, f1, (int)NGR, 1024, 256, 0, 1);
    dense(f1, 1024, fc2_w, fc2_b, f2, (int)NGR, 128, 1024, 0, 1);
    final_kernel<<<1, 256, 0, stream>>>(f2, out_w, out_b, out);
}
